// ResNetGAT_27582279975628
// MI455X (gfx1250) — matrix operand order repaired and checked by reading, not compiled, not run on hardware
//
#include <hip/hip_runtime.h>
#include <hip/hip_bf16.h>

typedef _Float16 half_t;
typedef __attribute__((ext_vector_type(16))) _Float16 v16h;
typedef __attribute__((ext_vector_type(8)))  _Float16 v8h;
typedef __attribute__((ext_vector_type(8)))  float    v8f;

#define NBATCH 4096
#define LSEQ   256
#define NROWS  (NBATCH * LSEQ)   // 1,048,576 (n*L rows)
#define SLOPE  0.01f

__device__ __forceinline__ float lrelu(float v) { return v > 0.f ? v : SLOPE * v; }

// ---------------------------------------------------------------------------
// Pack weights (or hidden^T for the attention GEMM) into WMMA B-register
// layout: per (tap, kchunk32, ntile16) a 512-half block, lane-major.
// B element for lane L, slot t: col = L&15, K = (L>>4)*16 + t   (f16 B 32x16)
// nck/nct are always powers of two -> passed as shift amounts (no SALU div).
// ---------------------------------------------------------------------------
__global__ void pack_wmma_b(const float* __restrict__ w, const float* __restrict__ g,
                            half_t* __restrict__ out,
                            int lg_nck, int lg_nct,
                            int cin_real, int cout_real,
                            int s_co, int s_ci, int s_k, int total) {
  int idx = blockIdx.x * 256 + threadIdx.x;
  if (idx >= total) return;
  int e    = idx & 511;
  int rest = idx >> 9;
  int nt = rest & ((1 << lg_nct) - 1); rest >>= lg_nct;
  int ck = rest & ((1 << lg_nck) - 1);
  int k  = rest >> lg_nck;
  int lane = e >> 4, t = e & 15;
  int col = lane & 15;
  int kk  = ((lane >> 4) << 4) + t;
  int ci  = ck * 32 + kk;
  int co  = nt * 16 + col;
  float val = 0.f;
  if (ci < cin_real && co < cout_real) {
    val = w[(size_t)co * s_co + (size_t)ci * s_ci + (size_t)k * s_k];
    if (g) val *= g[co];
  }
  out[idx] = (half_t)val;
}

// x [N, 12, 256] fp32  ->  act0 [N*L, 32] f16 channels-last, zero padded
__global__ void prep_input(const float* __restrict__ x, half_t* __restrict__ out) {
  size_t idx = (size_t)blockIdx.x * 256 + threadIdx.x;   // over NROWS*32
  int c = idx & 31;
  size_t r = idx >> 5;
  int l = (int)(r & 255);
  int n = (int)(r >> 8);
  float v = (c < 12) ? x[((size_t)n * 12 + c) * 256 + l] : 0.f;
  out[idx] = (half_t)v;
}

// ---------------------------------------------------------------------------
// Implicit-GEMM conv1d / dense GEMM with f16 WMMA, fp32 accumulate.
// Compile-time shape: NCT cout tiles, TAPS filter taps, NCK K-chunks of 32,
// LROW = per-sample row period (power of two: 256 conv / 4096 dense).
// Block = 8 waves; wave -> (sub rowblock, nt couttile). Stages
// (MEXT + TAPS - 1) x (NCK*32) f16 rows into LDS with zero halo at sample
// borders; taps/K loops fully unrolled so ds/global loads pipeline with WMMA.
// Epilogue: + bias, + optional residual (optionally per-channel affine),
// ReLU / LeakyReLU, stores f16 [rows][COUTP] and/or fp32 [rows][ldf].
// ---------------------------------------------------------------------------
template <int NCT, int TAPS, int NCK, int LROW>
__global__ __launch_bounds__(256) void conv_wmma_f16(
    const half_t* __restrict__ in, const half_t* __restrict__ wpk,
    const float* __restrict__ bias, const half_t* __restrict__ res,
    const float* __restrict__ res_g, const float* __restrict__ res_b,
    half_t* __restrict__ out_h, float* __restrict__ out_f,
    int act, int cout_real, int ldf) {
  constexpr int CINP  = NCK * 32;
  constexpr int COUTP = NCT * 16;
  constexpr int MSUB  = 8 / NCT;
  constexpr int MEXT  = MSUB * 16;
  constexpr int PAD   = (TAPS - 1) / 2;
  constexpr int TROWS = MEXT + TAPS - 1;
  constexpr int VPR   = CINP / 8;        // 16B vectors per row
  constexpr int TOTV  = TROWS * VPR;

  extern __shared__ char smem_raw[];
  half_t* lds = (half_t*)smem_raw;

  const int tid  = threadIdx.x;
  const int wave = tid >> 5;
  const int lane = tid & 31;
  const int r0   = blockIdx.x * MEXT;
  const int l0   = r0 & (LROW - 1);      // LROW is a power of two
  const int nbase = r0 - l0;

  // cooperative stage: rows [l0-PAD, l0+MEXT-1+PAD] x CINP
  for (int v = tid; v < TOTV; v += 256) {
    const int tr = v / VPR;              // constant divisor -> shifts/mads
    const int c8 = (v - tr * VPR) << 3;
    const int l  = l0 + tr - PAD;
    float4 val{0.f, 0.f, 0.f, 0.f};
    if (l >= 0 && l < LROW)
      val = *(const float4*)(in + (size_t)(nbase + l) * CINP + c8);
    *(float4*)(lds + (size_t)tr * CINP + c8) = val;
  }
  __syncthreads();

  const int nt    = wave % NCT;
  const int sub   = wave / NCT;
  const int mrow  = lane & 15;
  const int khalf = lane >> 4;

  v8f acc = {0.f, 0.f, 0.f, 0.f, 0.f, 0.f, 0.f, 0.f};

  constexpr int UNR = (NCK <= 8) ? NCK : 4;
#pragma unroll
  for (int k = 0; k < TAPS; ++k) {
    const half_t* arow0 = lds + (size_t)(sub * 16 + mrow + k) * CINP + khalf * 16;
    for (int ck0 = 0; ck0 < NCK; ck0 += UNR) {
#pragma unroll
      for (int u = 0; u < UNR; ++u) {
        const int ck = ck0 + u;
        const half_t* arow = arow0 + ck * 32;
        const half_t* bptr = wpk + ((size_t)(k * NCK + ck) * NCT + nt) * 512 + lane * 16;
        v8h lo = *(const v8h*)(arow);        // K = khalf*16 .. +7
        v8h hi = *(const v8h*)(arow + 8);    // K = khalf*16 + 8 .. +7
        v16h a = __builtin_shufflevector(lo, hi, 0, 1, 2, 3, 4, 5, 6, 7,
                                         8, 9, 10, 11, 12, 13, 14, 15);
        v16h b = *(const v16h*)bptr;
        if (ck + 1 < NCK) __builtin_prefetch(bptr + (size_t)NCT * 512, 0, 1);
        acc = __builtin_amdgcn_wmma_f32_16x16x32_f16(
            false, a, false, b, (short)0, acc, false, false);
      }
    }
  }

  const int col = lane & 15;
  const int co  = nt * 16 + col;
  const float bv = (bias != nullptr && co < cout_real) ? bias[co] : 0.f;
  const float rg = (res_g != nullptr && co < cout_real) ? res_g[co] : 1.f;
  const float rb = (res_b != nullptr && co < cout_real) ? res_b[co] : 0.f;

#pragma unroll
  for (int i = 0; i < 8; ++i) {
    const int r = r0 + sub * 16 + khalf * 8 + i;   // D layout: VGPR i -> M=i / M=8+i
    float vv = acc[i] + bv;
    if (res != nullptr) {
      float rv = (float)res[(size_t)r * COUTP + co];
      if (res_g != nullptr) rv = rv * rg + rb;
      vv += rv;
    }
    if (act == 1)      vv = vv > 0.f ? vv : 0.f;
    else if (act == 2) vv = lrelu(vv);
    if (out_h != nullptr) out_h[(size_t)r * COUTP + co] = (half_t)vv;
    if (out_f != nullptr && co < cout_real) out_f[(size_t)r * ldf + co] = vv;
  }
}

template <int NCT, int TAPS, int NCK, int LROW>
static void conv_call(hipStream_t stream, const half_t* in, const half_t* wpk,
                      const float* bias, const half_t* res, const float* rg,
                      const float* rb, half_t* oh, float* of,
                      int rows, int act, int coutr, int ldf) {
  constexpr int MEXT = (8 / NCT) * 16;
  constexpr size_t SH = (size_t)(MEXT + TAPS - 1) * (NCK * 32) * sizeof(half_t);
  if (SH > 64 * 1024)
    (void)hipFuncSetAttribute(
        reinterpret_cast<const void*>(&conv_wmma_f16<NCT, TAPS, NCK, LROW>),
        hipFuncAttributeMaxDynamicSharedMemorySize, 160 * 1024);
  conv_wmma_f16<NCT, TAPS, NCK, LROW><<<rows / MEXT, 256, SH, stream>>>(
      in, wpk, bias, res, rg, rb, oh, of, act, coutr, ldf);
}

// Global average pool over L: h3 [N*L,128] f16 -> hidden fp32 + f16
__global__ void gap_kernel(const half_t* __restrict__ h3,
                           float* __restrict__ hid_f, half_t* __restrict__ hid_h) {
  int n = blockIdx.x, d = threadIdx.x;          // 128 threads
  const half_t* p = h3 + (size_t)n * LSEQ * 128 + d;
  float s = 0.f;
  for (int l = 0; l < LSEQ; ++l) s += (float)p[(size_t)l * 128];
  s *= (1.f / (float)LSEQ);
  hid_f[n * 128 + d] = s;
  hid_h[n * 128 + d] = (half_t)s;
}

// s_src[n] = x1[n,:].a[:128], s_dst[n] = x1[n,:].a[128:]
__global__ void attn_scores(const float* __restrict__ x1f, const float* __restrict__ a,
                            float* __restrict__ s_src, float* __restrict__ s_dst) {
  __shared__ float s1[128];
  __shared__ float s2[128];
  int n = blockIdx.x, d = threadIdx.x;          // 128 threads
  float v = x1f[n * 128 + d];
  s1[d] = v * a[d];
  s2[d] = v * a[128 + d];
  __syncthreads();
  for (int s = 64; s > 0; s >>= 1) {
    if (d < s) { s1[d] += s1[d + s]; s2[d] += s2[d + s]; }
    __syncthreads();
  }
  if (d == 0) { s_src[n] = s1[0]; s_dst[n] = s2[0]; }
}

__global__ void max_reduce(const float* __restrict__ s, float* __restrict__ out) {
  __shared__ float sm[256];
  float m = -3.4e38f;
  for (int j = threadIdx.x; j < NBATCH; j += 256) m = fmaxf(m, s[j]);
  sm[threadIdx.x] = m;
  __syncthreads();
  for (int w = 128; w > 0; w >>= 1) {
    if (threadIdx.x < w) sm[threadIdx.x] = fmaxf(sm[threadIdx.x], sm[threadIdx.x + w]);
    __syncthreads();
  }
  if (threadIdx.x == 0) *out = sm[0];
}

// Z_i = sum_j exp(leaky(sd_i + ss_j) - m_i); m_i = leaky(sd_i + max_j ss_j)
__global__ void att_z(const float* __restrict__ s_src, const float* __restrict__ s_dst,
                      const float* __restrict__ ssmax, float* __restrict__ Z) {
  __shared__ float sm[256];
  int i = blockIdx.x;
  float sd = s_dst[i];
  float m = lrelu(sd + *ssmax);
  float z = 0.f;
  for (int j = threadIdx.x; j < NBATCH; j += 256)
    z += __expf(lrelu(sd + s_src[j]) - m);
  sm[threadIdx.x] = z;
  __syncthreads();
  for (int w = 128; w > 0; w >>= 1) {
    if (threadIdx.x < w) sm[threadIdx.x] += sm[threadIdx.x + w];
    __syncthreads();
  }
  if (threadIdx.x == 0) Z[i] = sm[0];
}

// att[i,j] materialized as f16 (L2-resident, 32MB)
__global__ void att_p(const float* __restrict__ s_src, const float* __restrict__ s_dst,
                      const float* __restrict__ ssmax, const float* __restrict__ Z,
                      half_t* __restrict__ P) {
  int i = blockIdx.x;
  float sd = s_dst[i];
  float m = lrelu(sd + *ssmax);
  float zi = 1.f / Z[i];
  for (int j = threadIdx.x; j < NBATCH; j += 256)
    P[(size_t)i * NBATCH + j] = (half_t)(__expf(lrelu(sd + s_src[j]) - m) * zi);
}

// ---------------------------------------------------------------------------
extern "C" void kernel_launch(void* const* d_in, const int* in_sizes, int n_in,
                              void* d_out, int out_size, void* d_ws, size_t ws_size,
                              hipStream_t stream) {
  (void)in_sizes; (void)n_in; (void)out_size; (void)ws_size;

  const float* x    = (const float*)d_in[0];
  const float* r1w1 = (const float*)d_in[1];
  const float* r1g1 = (const float*)d_in[2];
  const float* r1b1 = (const float*)d_in[3];
  const float* r1w2 = (const float*)d_in[4];
  const float* r1g2 = (const float*)d_in[5];
  const float* r1b2 = (const float*)d_in[6];
  const float* r1w3 = (const float*)d_in[7];
  const float* r1g3 = (const float*)d_in[8];
  const float* r1b3 = (const float*)d_in[9];
  const float* r1ws = (const float*)d_in[10];
  const float* r1gs = (const float*)d_in[11];
  const float* r1bs = (const float*)d_in[12];
  const float* r2w1 = (const float*)d_in[13];
  const float* r2g1 = (const float*)d_in[14];
  const float* r2b1 = (const float*)d_in[15];
  const float* r2w2 = (const float*)d_in[16];
  const float* r2g2 = (const float*)d_in[17];
  const float* r2b2 = (const float*)d_in[18];
  const float* r2w3 = (const float*)d_in[19];
  const float* r2g3 = (const float*)d_in[20];
  const float* r2b3 = (const float*)d_in[21];
  const float* r2ws = (const float*)d_in[22];
  const float* r2gs = (const float*)d_in[23];
  const float* r2bs = (const float*)d_in[24];
  const float* r3w1 = (const float*)d_in[25];
  const float* r3g1 = (const float*)d_in[26];
  const float* r3b1 = (const float*)d_in[27];
  const float* r3w2 = (const float*)d_in[28];
  const float* r3g2 = (const float*)d_in[29];
  const float* r3b2 = (const float*)d_in[30];
  const float* r3w3 = (const float*)d_in[31];
  const float* r3g3 = (const float*)d_in[32];
  const float* r3b3 = (const float*)d_in[33];
  const float* r3gs = (const float*)d_in[34];
  const float* r3bs = (const float*)d_in[35];
  const float* fcw  = (const float*)d_in[36];
  const float* fcb  = (const float*)d_in[37];
  const float* trw  = (const float*)d_in[38];
  const float* trb  = (const float*)d_in[39];
  const float* avec = (const float*)d_in[40];
  const float* g1w  = (const float*)d_in[41];
  const float* g1b  = (const float*)d_in[42];
  const float* g2w  = (const float*)d_in[43];
  const float* g2b  = (const float*)d_in[44];

  // workspace carve (256B aligned bump allocator)
  char* base = (char*)d_ws;
  size_t off = 0;
  auto carve = [&](size_t bytes) -> void* {
    void* p = base + off;
    off += (bytes + 255) & ~(size_t)255;
    return p;
  };
  half_t* act0 = (half_t*)carve((size_t)NROWS * 32 * 2);
  half_t* bufA = (half_t*)carve((size_t)NROWS * 128 * 2);
  half_t* bufB = (half_t*)carve((size_t)NROWS * 128 * 2);
  half_t* bufS = (half_t*)carve((size_t)NROWS * 128 * 2);
  half_t* Patt = (half_t*)carve((size_t)NBATCH * NBATCH * 2);
  float*  hidf = (float*)carve((size_t)NBATCH * 128 * 4);
  half_t* hidh = (half_t*)carve((size_t)NBATCH * 128 * 2);
  float*  x1f  = (float*)carve((size_t)NBATCH * 128 * 4);
  half_t* h2h  = (half_t*)carve((size_t)NBATCH * 128 * 2);
  half_t* gh   = (half_t*)carve((size_t)NBATCH * 128 * 2);
  float*  ssrc = (float*)carve(NBATCH * 4);
  float*  sdst = (float*)carve(NBATCH * 4);
  float*  Zrow = (float*)carve(NBATCH * 4);
  float*  smax = (float*)carve(256);
  auto carveW = [&](int taps, int nck, int nct) -> half_t* {
    return (half_t*)carve((size_t)taps * nck * nct * 512 * 2);
  };
  half_t* pk_r1s  = carveW(1, 1, 4);
  half_t* pk_r1c1 = carveW(7, 1, 4);
  half_t* pk_r1c2 = carveW(5, 2, 4);
  half_t* pk_r1c3 = carveW(3, 2, 4);
  half_t* pk_r2s  = carveW(1, 2, 8);
  half_t* pk_r2c1 = carveW(7, 2, 8);
  half_t* pk_r2c2 = carveW(5, 4, 8);
  half_t* pk_r2c3 = carveW(3, 4, 8);
  half_t* pk_r3c1 = carveW(7, 4, 8);
  half_t* pk_r3c2 = carveW(5, 4, 8);
  half_t* pk_r3c3 = carveW(3, 4, 8);
  half_t* pk_fc   = carveW(1, 4, 1);
  half_t* pk_tr   = carveW(1, 4, 8);
  half_t* pk_g1   = carveW(1, 4, 8);
  half_t* pk_g2   = carveW(1, 4, 1);
  half_t* pk_hB   = carveW(1, 128, 8);

  auto packW = [&](const float* w, const float* g, half_t* out, int taps, int nck,
                   int nct, int cin, int cout, int sco, int sci, int sk) {
    int total = taps * nck * nct * 512;
    pack_wmma_b<<<(total + 255) / 256, 256, 0, stream>>>(
        w, g, out, __builtin_ctz(nck), __builtin_ctz(nct),
        cin, cout, sco, sci, sk, total);
  };

  // --- weight packing (BN scale g folded into weights) ---
  packW(r1ws, r1gs, pk_r1s,  1, 1, 4,  12,  64,  12, 1, 1);
  packW(r1w1, r1g1, pk_r1c1, 7, 1, 4,  12,  64,  84, 7, 1);
  packW(r1w2, r1g2, pk_r1c2, 5, 2, 4,  64,  64, 320, 5, 1);
  packW(r1w3, r1g3, pk_r1c3, 3, 2, 4,  64,  64, 192, 3, 1);
  packW(r2ws, r2gs, pk_r2s,  1, 2, 8,  64, 128,  64, 1, 1);
  packW(r2w1, r2g1, pk_r2c1, 7, 2, 8,  64, 128, 448, 7, 1);
  packW(r2w2, r2g2, pk_r2c2, 5, 4, 8, 128, 128, 640, 5, 1);
  packW(r2w3, r2g3, pk_r2c3, 3, 4, 8, 128, 128, 384, 3, 1);
  packW(r3w1, r3g1, pk_r3c1, 7, 4, 8, 128, 128, 896, 7, 1);
  packW(r3w2, r3g2, pk_r3c2, 5, 4, 8, 128, 128, 640, 5, 1);
  packW(r3w3, r3g3, pk_r3c3, 3, 4, 8, 128, 128, 384, 3, 1);
  packW(fcw, nullptr, pk_fc, 1, 4, 1, 128,   8, 128, 1, 1);
  packW(trw, nullptr, pk_tr, 1, 4, 8, 128, 128, 128, 1, 1);
  packW(g1w, nullptr, pk_g1, 1, 4, 8, 128, 128, 128, 1, 1);
  packW(g2w, nullptr, pk_g2, 1, 4, 1, 128,   8, 128, 1, 1);

  prep_input<<<(NROWS * 32) / 256, 256, 0, stream>>>(x, act0);

  // --- resblock 1 (12->64) ---
  conv_call<4, 1, 1, LSEQ>(stream, act0, pk_r1s,  r1bs, nullptr, nullptr, nullptr,
                           bufS, nullptr, NROWS, 0, 64, 0);              // shortcut
  conv_call<4, 7, 1, LSEQ>(stream, act0, pk_r1c1, r1b1, nullptr, nullptr, nullptr,
                           bufA, nullptr, NROWS, 1, 64, 0);
  conv_call<4, 5, 2, LSEQ>(stream, bufA, pk_r1c2, r1b2, nullptr, nullptr, nullptr,
                           bufB, nullptr, NROWS, 1, 64, 0);
  conv_call<4, 3, 2, LSEQ>(stream, bufB, pk_r1c3, r1b3, bufS, nullptr, nullptr,
                           bufA, nullptr, NROWS, 1, 64, 0);              // +sc, relu
  // --- resblock 2 (64->128), h1 in bufA ---
  conv_call<8, 1, 2, LSEQ>(stream, bufA, pk_r2s,  r2bs, nullptr, nullptr, nullptr,
                           bufS, nullptr, NROWS, 0, 128, 0);
  conv_call<8, 7, 2, LSEQ>(stream, bufA, pk_r2c1, r2b1, nullptr, nullptr, nullptr,
                           bufB, nullptr, NROWS, 1, 128, 0);
  conv_call<8, 5, 4, LSEQ>(stream, bufB, pk_r2c2, r2b2, nullptr, nullptr, nullptr,
                           bufA, nullptr, NROWS, 1, 128, 0);
  conv_call<8, 3, 4, LSEQ>(stream, bufA, pk_r2c3, r2b3, bufS, nullptr, nullptr,
                           bufB, nullptr, NROWS, 1, 128, 0);
  // --- resblock 3 (128->128, BN-only shortcut), h2 in bufB ---
  conv_call<8, 7, 4, LSEQ>(stream, bufB, pk_r3c1, r3b1, nullptr, nullptr, nullptr,
                           bufA, nullptr, NROWS, 1, 128, 0);
  conv_call<8, 5, 4, LSEQ>(stream, bufA, pk_r3c2, r3b2, nullptr, nullptr, nullptr,
                           bufS, nullptr, NROWS, 1, 128, 0);
  conv_call<8, 3, 4, LSEQ>(stream, bufS, pk_r3c3, r3b3, bufB, r3gs, r3bs,
                           bufA, nullptr, NROWS, 1, 128, 0);             // +gs*x+bs

  gap_kernel<<<NBATCH, 128, 0, stream>>>(bufA, hidf, hidh);

  float* origin = (float*)d_out;
  float* gat    = origin + NBATCH * 8;
  float* h2f    = gat + NBATCH * 8;

  // origin_out = hidden @ fc_w.T + fc_b
  conv_call<1, 1, 4, NBATCH>(stream, hidh, pk_fc, fcb, nullptr, nullptr, nullptr,
                             nullptr, origin, NBATCH, 0, 8, 8);
  // x1 = hidden @ tr_w.T + tr_b
  conv_call<8, 1, 4, NBATCH>(stream, hidh, pk_tr, trb, nullptr, nullptr, nullptr,
                             nullptr, x1f, NBATCH, 0, 128, 128);

  attn_scores<<<NBATCH, 128, 0, stream>>>(x1f, avec, ssrc, sdst);
  max_reduce<<<1, 256, 0, stream>>>(ssrc, smax);
  att_z<<<NBATCH, 256, 0, stream>>>(ssrc, sdst, smax, Zrow);
  att_p<<<NBATCH, 256, 0, stream>>>(ssrc, sdst, smax, Zrow, Patt);

  // pack hidden^T as WMMA-B for the attention GEMM
  packW(hidf, nullptr, pk_hB, 1, 128, 8, NBATCH, 128, 1, 128, 1);
  // hidden2 = att @ hidden + hidden   (M=4096, N=128, K=4096; 128KB LDS A-tile)
  conv_call<8, 1, 128, NBATCH>(stream, Patt, pk_hB, nullptr, hidh, nullptr, nullptr,
                               h2h, h2f, NBATCH, 0, 128, 128);
  // g = leaky(hidden2 @ g1_w.T + g1_b)
  conv_call<8, 1, 4, NBATCH>(stream, h2h, pk_g1, g1b, nullptr, nullptr, nullptr,
                             gh, nullptr, NBATCH, 2, 128, 0);
  // gat_out = g @ g2_w.T + g2_b
  conv_call<1, 1, 4, NBATCH>(stream, gh, pk_g2, g2b, nullptr, nullptr, nullptr,
                             nullptr, gat, NBATCH, 0, 8, 8);
}
